// Matching_1331439862516
// MI455X (gfx1250) — compile-verified, matrix operand order checked by reading, not run on hardware
//
#include <hip/hip_runtime.h>

// ---------------------------------------------------------------------------
// Matching module (flow upsample -> warp -> 7x7 correlation -> 3 convs -> flow
// head) for MI455X / gfx1250.  GEMM work on v_wmma_f32_16x16x32_f16 (wave32).
// All f16 activations are NHWC (channels innermost) so LDS staging is pure
// contiguous 16B copies -> GLOBAL_LOAD_ASYNC_TO_LDS_B128 (ASYNCcnt-tracked).
// ---------------------------------------------------------------------------

typedef __attribute__((ext_vector_type(16))) _Float16 v16h;
typedef __attribute__((ext_vector_type(8)))  float    v8f;
typedef __attribute__((ext_vector_type(4)))  unsigned int u32x4;
typedef int gv4i __attribute__((vector_size(16)));   // matches builtin param

static constexpr int Bn = 4, Cn = 128, Hn = 192, Wn = 320;
static constexpr float MULTV = 2.5f;   // 20 / 2^(5-2)

#define WMMA_F16(a, b, c) \
  __builtin_amdgcn_wmma_f32_16x16x32_f16(false, (a), false, (b), (short)0, (c), false, false)

#if __has_builtin(__builtin_amdgcn_global_load_async_to_lds_b128)
#define HAVE_ASYNC_LD 1
#else
#define HAVE_ASYNC_LD 0
#endif
#if __has_builtin(__builtin_amdgcn_global_store_async_from_lds_b128)
#define HAVE_ASYNC_ST 1
#else
#define HAVE_ASYNC_ST 0
#endif

#define AS1_V4I __attribute__((address_space(1))) gv4i*
#define AS3_V4I __attribute__((address_space(3))) gv4i*

// 16B global -> LDS copy (async engine, ASYNCcnt-tracked)
static __device__ __forceinline__ void cp_g2l_b128(_Float16* ldst, const _Float16* gsrc) {
#if HAVE_ASYNC_LD
  __builtin_amdgcn_global_load_async_to_lds_b128(
      (AS1_V4I)(gv4i*)gsrc, (AS3_V4I)(gv4i*)ldst, 0, 0);
#else
  *(u32x4*)ldst = *(const u32x4*)gsrc;
#endif
}

// 16B LDS -> global copy
static __device__ __forceinline__ void cp_l2g_b128(_Float16* gdst, const _Float16* lsrc) {
#if HAVE_ASYNC_ST
  __builtin_amdgcn_global_store_async_from_lds_b128(
      (AS1_V4I)(gv4i*)gdst, (AS3_V4I)(gv4i*)lsrc, 0, 0);
#else
  *(u32x4*)gdst = *(const u32x4*)lsrc;
#endif
}

static __device__ __forceinline__ void async_join() {
#if HAVE_ASYNC_LD || HAVE_ASYNC_ST
#if __has_builtin(__builtin_amdgcn_s_wait_asynccnt)
  __builtin_amdgcn_s_wait_asynccnt(0);
#else
  asm volatile("s_wait_asynccnt 0" ::: "memory");
#endif
#endif
}

// WMMA fragment load from LDS laid out [rc][c] with padded stride (halves).
// Lane needs halves {cb+8*hi .. +7} and {cb+8*hi+16 .. +23}: two aligned
// 16-byte ds_load_b128.  Padded stride (multiple of 8, not of 32) spreads
// banks without any swizzle.
static __device__ __forceinline__ v16h load_frag(const _Float16* lds, int rc,
                                                 int stride, int cbase) {
  const _Float16* p = lds + rc * stride + cbase;
  union { u32x4 q[2]; v16h v; } u;
  u.q[0] = *(const u32x4*)p;
  u.q[1] = *(const u32x4*)(p + 16);
  return u.v;
}

// ---------------------------------------------------------------------------
// 1) flow upsample: transposed conv 4x4, lhs_dilation=2, pad 2, groups=2.
// ---------------------------------------------------------------------------
__global__ void upflow_kernel(const float* __restrict__ flow,
                              const float* __restrict__ up_w,
                              float* __restrict__ flow_up) {
  int i = blockIdx.x * blockDim.x + threadIdx.x;
  const int total = Bn * 2 * Hn * Wn;
  if (i >= total) return;
  int x = i % Wn; int t = i / Wn;
  int y = t % Hn; t /= Hn;
  int g = t & 1;  int b = t >> 1;
  const int Hi = Hn / 2, Wi = Wn / 2;
  const float* in = flow + (size_t)(b * 2 + g) * Hi * Wi;
  float acc = 0.f;
  for (int ky = 0; ky < 4; ++ky) {
    int py = y + ky - 2;
    if (py & 1) continue;
    int iy = py >> 1;
    if (iy < 0 || iy >= Hi) continue;
    for (int kx = 0; kx < 4; ++kx) {
      int px = x + kx - 2;
      if (px & 1) continue;
      int ix = px >> 1;
      if (ix < 0 || ix >= Wi) continue;
      acc += up_w[g * 16 + (3 - ky) * 4 + (3 - kx)] * in[iy * Wi + ix];
    }
  }
  flow_up[i] = acc;
}

// ---------------------------------------------------------------------------
// 2) f1 fp32 NCHW -> f16 NHWC
// ---------------------------------------------------------------------------
__global__ void f1cvt_kernel(const float* __restrict__ feats,
                             _Float16* __restrict__ f1h) {
  int pix = blockIdx.x * blockDim.x + threadIdx.x;
  if (pix >= Bn * Hn * Wn) return;
  const size_t plane = (size_t)Hn * Wn;
  int b = pix / (Hn * Wn);
  int hw = pix - b * (Hn * Wn);
  const float* src = feats + (size_t)(b * 2) * Cn * plane + hw;  // feats[:,0]
  _Float16* dst = f1h + (size_t)pix * Cn;
  union { _Float16 h[8]; u32x4 q; } pk;
  for (int c8 = 0; c8 < Cn / 8; ++c8) {
    for (int j = 0; j < 8; ++j)
      pk.h[j] = (_Float16)src[(size_t)(c8 * 8 + j) * plane];
    *(u32x4*)(dst + c8 * 8) = pk.q;
  }
}

// ---------------------------------------------------------------------------
// 3) bilinear warp of feats[:,1] by flow_up*MULT with OOB mask -> f16 NHWC
// ---------------------------------------------------------------------------
__global__ void warp_kernel(const float* __restrict__ feats,
                            const float* __restrict__ flow_up,
                            _Float16* __restrict__ out) {
  int i = blockIdx.x * blockDim.x + threadIdx.x;
  if (i >= Bn * Hn * Wn) return;
  int x = i % Wn; int t = i / Wn;
  int y = t % Hn; int b = t / Hn;
  const size_t plane = (size_t)Hn * Wn;
  const float* img = feats + (size_t)(b * 2 + 1) * Cn * plane;  // f2, NCHW
  float fx = flow_up[((size_t)(b * 2) + 0) * plane + (size_t)y * Wn + x];
  float fy = flow_up[((size_t)(b * 2) + 1) * plane + (size_t)y * Wn + x];
  float px = (float)x + fx * MULTV;
  float py = (float)y + fy * MULTV;
  float pxc = fminf(fmaxf(px, 0.f), (float)(Wn - 1));
  float pyc = fminf(fmaxf(py, 0.f), (float)(Hn - 1));
  float x0f = floorf(pxc), y0f = floorf(pyc);
  float tx = pxc - x0f, ty = pyc - y0f;
  int x0 = (int)x0f, y0 = (int)y0f;
  int x1 = x0 + 1 < Wn - 1 ? x0 + 1 : Wn - 1;
  int y1 = y0 + 1 < Hn - 1 ? y0 + 1 : Hn - 1;
  float fx0 = floorf(px), fy0 = floorf(py);
  float mtx = px - fx0, mty = py - fy0;
  float mm = 0.f;
  for (int dy = 0; dy < 2; ++dy)
    for (int dx = 0; dx < 2; ++dx) {
      float xi = fx0 + (float)dx, yi = fy0 + (float)dy;
      float wgt = (dx ? mtx : 1.f - mtx) * (dy ? mty : 1.f - mty);
      if (xi >= 0.f && xi <= (float)(Wn - 1) && yi >= 0.f && yi <= (float)(Hn - 1))
        mm += wgt;
    }
  float mask = (mm >= 1.0f) ? 1.f : 0.f;
  float w00 = (1.f - tx) * (1.f - ty), w01 = tx * (1.f - ty);
  float w10 = (1.f - tx) * ty,         w11 = tx * ty;
  size_t o00 = (size_t)y0 * Wn + x0, o01 = (size_t)y0 * Wn + x1;
  size_t o10 = (size_t)y1 * Wn + x0, o11 = (size_t)y1 * Wn + x1;
  _Float16* op = out + (size_t)i * Cn;  // NHWC row for this pixel
  union { _Float16 h[8]; u32x4 q; } pk;
  for (int c8 = 0; c8 < Cn / 8; ++c8) {
    for (int j = 0; j < 8; ++j) {
      const float* ip = img + (size_t)(c8 * 8 + j) * plane;
      float v = ip[o00] * w00 + ip[o01] * w01 + ip[o10] * w10 + ip[o11] * w11;
      pk.h[j] = (_Float16)(v * mask);
    }
    *(u32x4*)(op + c8 * 8) = pk.q;
  }
}

// ---------------------------------------------------------------------------
// 4) weight pack: OIHW fp32 -> f16 B-fragments in exact WMMA lane order.
// k = tap*CINP + c.  One lane's fragment = one contiguous 32B load.
// ---------------------------------------------------------------------------
__global__ void pack_w_kernel(const float* __restrict__ w,
                              _Float16* __restrict__ wp,
                              int COUT, int CIN, int CINP) {
  int t = blockIdx.x * blockDim.x + threadIdx.x;
  const int NT = COUT / 16;
  const int KC = CINP / 32;
  const int NKT = KC * 9;
  const int total = NKT * NT * 32;
  if (t >= total) return;
  int lane = t & 31;
  int q = t >> 5;
  int nt = q % NT;
  int kt = q / NT;
  int tap = kt / KC;
  int cb = (kt - tap * KC) * 32;
  int ky = tap / 3, kx = tap - ky * 3;
  int n = nt * 16 + (lane & 15);
  int hi = (lane >> 4) * 8;
  _Float16* dst = wp + ((size_t)(kt * NT + nt) * 32 + lane) * 16;
  for (int e = 0; e < 16; ++e) {
    int klocal = (e < 8 ? e : e + 8) + hi;
    int c = cb + klocal;
    float v = 0.f;
    if (c < CIN) v = w[(((size_t)n * CIN + c) * 3 + ky) * 3 + kx];
    dst[e] = (_Float16)v;
  }
}

// ---------------------------------------------------------------------------
// 5) correlation via banded WMMA GEMM (2 waves / 16-pixel row tile).
// T[m,j] = sum_c f1[px+m,c] * f2w[y+di-3, x0-3+j, c], two N=16 tiles (j<32),
// band extract corr[m, di*7+dj] = leaky(T[m, m+dj])/128 -> NHWC 64ch f16
// (channels 49..63 zeroed so conv1's K is a multiple of 32).
// ---------------------------------------------------------------------------
__global__ __launch_bounds__(64)
void corr_wmma(const _Float16* __restrict__ f1,
               const _Float16* __restrict__ f2,
               _Float16* __restrict__ corr) {
  constexpr int STR = Cn + 8;                  // padded LDS stride (halves)
  __shared__ _Float16 l2[224 * STR];           // 60928 B: 7 rows x 32 cols
  __shared__ _Float16 l1buf[16 * STR];         //  4352 B: f1 tile, then T
  float* T = (float*)l1buf;                    // aliased after A-frag preload
  const int x0 = blockIdx.x * 16;
  const int y = blockIdx.y;
  const int b = blockIdx.z;
  const size_t plane = (size_t)Hn * Wn;
  const _Float16* f1b = f1 + (size_t)b * plane * Cn;
  const _Float16* f2b = f2 + (size_t)b * plane * Cn;
  const int tid = threadIdx.x, lane = tid & 31, wv = tid >> 5;

  // stage f2 tile: rows y-3..y+3, cols x0-3..x0+28 (async 16B copies)
  for (int g = tid; g < 224 * (Cn / 8); g += 64) {
    int rc = g >> 4, cg = g & 15;
    int row = rc >> 5, col = rc & 31;
    int gy = y + row - 3, gx = x0 + col - 3;
    _Float16* dst = l2 + rc * STR + cg * 8;
    if (gy >= 0 && gy < Hn && gx >= 0 && gx < Wn)
      cp_g2l_b128(dst, f2b + ((size_t)(gy * Wn + gx) * Cn + cg * 8));
    else {
      u32x4 z = {0u, 0u, 0u, 0u};
      *(u32x4*)dst = z;
    }
  }
  // stage f1 16-pixel block (always in range)
  for (int g = tid; g < 16 * (Cn / 8); g += 64) {
    int col = g >> 4, cg = g & 15;
    cp_g2l_b128(l1buf + col * STR + cg * 8,
                f1b + ((size_t)(y * Wn + x0 + col) * Cn + cg * 8));
  }
  // zero padded corr channels 49..63 (NHWC)
  for (int g = tid; g < 15 * 16; g += 64) {
    int ch = 49 + g / 16; int m = g % 16;
    corr[((size_t)((b * Hn + y) * Wn + x0 + m)) * 64 + ch] = (_Float16)0.f;
  }
  async_join();
  __syncthreads();

  const int m_or = (lane >> 4) << 3;
  v16h afr[4];
  for (int kt = 0; kt < 4; ++kt)
    afr[kt] = load_frag(l1buf, lane & 15, STR, kt * 32 + m_or);
  __syncthreads();   // protect T alias over l1buf

  const float inv = 1.0f / 128.0f;
  for (int di = 0; di < 7; ++di) {
    // batch-issue all 8 ds_load_b128 for this di, then the 4-WMMA chain
    const int rcB = di * 32 + wv * 16 + (lane & 15);
    v16h bfr[4];
    for (int kt = 0; kt < 4; ++kt)
      bfr[kt] = load_frag(l2, rcB, STR, kt * 32 + m_or);
    v8f acc = {};
    for (int kt = 0; kt < 4; ++kt)
      acc = WMMA_F16(afr[kt], bfr[kt], acc);
    for (int r = 0; r < 8; ++r)
      T[(r + m_or) * 32 + wv * 16 + (lane & 15)] = acc[r];
    __syncthreads();
    for (int o = tid; o < 112; o += 64) {
      int m = o / 7, dj = o - m * 7;
      float v = T[m * 32 + m + dj];          // col = m + dj (band extract)
      v = v > 0.f ? v : 0.1f * v;            // leaky_relu(., 0.1) / C
      corr[((size_t)((b * Hn + y) * Wn + x0 + m)) * 64 + di * 7 + dj] =
          (_Float16)(v * inv);
    }
    __syncthreads();
  }
}

// ---------------------------------------------------------------------------
// 6) 3x3 conv, implicit GEMM on WMMA, NHWC in/out.  (COUT/16) waves share one
// async-staged LDS tile [3 rows][18 cols][CINP]; 2-stage software pipeline on
// the K loop; epilogue transposes through LDS and streams out with async
// LDS->global 16B stores.
// ---------------------------------------------------------------------------
template <int CINP, int COUT>
__global__ void conv3x3_wmma(const _Float16* __restrict__ in,
                             const _Float16* __restrict__ wp,
                             const float* __restrict__ bias,
                             _Float16* __restrict__ out) {
  constexpr int NT = COUT / 16;
  constexpr int KC = CINP / 32;
  constexpr int NKT = KC * 9;
  constexpr int STR = CINP + 8;
  __shared__ _Float16 tile[54 * STR];
  __shared__ _Float16 out_t[16 * COUT];
  const int x0 = blockIdx.x * 16;
  const int y = blockIdx.y;
  const int b = blockIdx.z;
  const int tid = threadIdx.x, lane = tid & 31, wv = tid >> 5;
  const size_t plane = (size_t)Hn * Wn;
  const _Float16* inb = in + (size_t)b * plane * CINP;

  for (int g = tid; g < 54 * (CINP / 8); g += NT * 32) {
    int rc = g / (CINP / 8);
    int cg = g - rc * (CINP / 8);
    int row = rc / 18, col = rc - row * 18;
    int gy = y + row - 1, gx = x0 + col - 1;
    _Float16* dst = tile + rc * STR + cg * 8;
    if (gy >= 0 && gy < Hn && gx >= 0 && gx < Wn)
      cp_g2l_b128(dst, inb + ((size_t)(gy * Wn + gx) * CINP + cg * 8));
    else {
      u32x4 z = {0u, 0u, 0u, 0u};
      *(u32x4*)dst = z;
    }
  }
  async_join();
  __syncthreads();

  const int m_or = (lane >> 4) << 3;
  auto a_frag = [&](int kt) -> v16h {
    int tap = kt / KC;
    int cb = (kt - tap * KC) * 32;
    int ky = tap / 3, kx = tap - ky * 3;
    int rcA = ky * 18 + (lane & 15) + kx;
    return load_frag(tile, rcA, STR, cb + m_or);
  };
  auto b_frag = [&](int kt) -> v16h {
    return *(const v16h*)(wp + ((size_t)(kt * NT + wv) * 32 + lane) * 16);
  };

  v8f acc = {};
  v16h a_cur = a_frag(0);
  v16h b_cur = b_frag(0);
  for (int kt = 0; kt < NKT; ++kt) {
    v16h a_nxt = a_cur, b_nxt = b_cur;
    if (kt + 1 < NKT) {          // prefetch next fragments while WMMA runs
      a_nxt = a_frag(kt + 1);
      b_nxt = b_frag(kt + 1);
    }
    acc = WMMA_F16(a_cur, b_cur, acc);
    a_cur = a_nxt;
    b_cur = b_nxt;
  }

  // bias + leaky, transpose through LDS to channel-contiguous rows
  const int n = lane & 15;
  const int ch = wv * 16 + n;
  const float bs = bias[ch];
  for (int r = 0; r < 8; ++r) {
    float v = acc[r] + bs;
    v = v > 0.f ? v : 0.1f * v;
    out_t[(r + m_or) * COUT + ch] = (_Float16)v;
  }
  __syncthreads();

  for (int g = tid; g < 16 * (COUT / 8); g += NT * 32) {
    int m = g / (COUT / 8);
    int cg = g - m * (COUT / 8);
    cp_l2g_b128(out + ((size_t)((b * Hn + y) * Wn + x0 + m) * COUT + cg * 8),
                out_t + m * COUT + cg * 8);
  }
  async_join();
}

// ---------------------------------------------------------------------------
// 7) 5x5 conv 32->2 (no activation) + bias + residual flow_up -> fp32 output
// ---------------------------------------------------------------------------
__global__ void conv4_flow(const _Float16* __restrict__ h3,
                           const float* __restrict__ w4,
                           const float* __restrict__ b4,
                           const float* __restrict__ flow_up,
                           float* __restrict__ out) {
  __shared__ float wl[1600];
  const int tid = threadIdx.x;
  for (int i = tid; i < 1600; i += blockDim.x) wl[i] = w4[i];
  __syncthreads();
  int gi = blockIdx.x * blockDim.x + tid;
  if (gi >= Bn * Hn * Wn) return;
  int x = gi % Wn; int t = gi / Wn;
  int y = t % Hn;  int b = t / Hn;
  const size_t plane = (size_t)Hn * Wn;
  const _Float16* hb = h3 + (size_t)b * plane * 32;   // NHWC 32ch
  float a0 = b4[0], a1 = b4[1];
  for (int ky = 0; ky < 5; ++ky) {
    int gy = y + ky - 2;
    if (gy < 0 || gy >= Hn) continue;
    for (int kx = 0; kx < 5; ++kx) {
      int gx = x + kx - 2;
      if (gx < 0 || gx >= Wn) continue;
      int tap = ky * 5 + kx;
      const _Float16* p = hb + (size_t)(gy * Wn + gx) * 32;
      for (int c = 0; c < 32; ++c) {
        float v = (float)p[c];
        a0 += v * wl[c * 25 + tap];
        a1 += v * wl[800 + c * 25 + tap];
      }
    }
  }
  size_t o = (size_t)(b * 2) * plane + (size_t)y * Wn + x;
  out[o] = flow_up[o] + a0;
  out[o + plane] = flow_up[o + plane] + a1;
}

// ---------------------------------------------------------------------------
extern "C" void kernel_launch(void* const* d_in, const int* in_sizes, int n_in,
                              void* d_out, int out_size, void* d_ws, size_t ws_size,
                              hipStream_t stream) {
  const float* feats = (const float*)d_in[0];
  const float* flow  = (const float*)d_in[1];
  const float* up_w  = (const float*)d_in[2];
  const float* w1 = (const float*)d_in[3];
  const float* b1 = (const float*)d_in[4];
  const float* w2 = (const float*)d_in[5];
  const float* b2 = (const float*)d_in[6];
  const float* w3 = (const float*)d_in[7];
  const float* b3 = (const float*)d_in[8];
  const float* w4 = (const float*)d_in[9];
  const float* b4 = (const float*)d_in[10];
  float* out = (float*)d_out;

  char* ws = (char*)d_ws;
  // byte offsets (all multiples of 256)
  float*    flow_up = (float*)(ws + 0);              //  1,966,080 B
  _Float16* f1h   = (_Float16*)(ws + 1966080);       // 62,914,560 B (NHWC128)
  _Float16* f2wh  = (_Float16*)(ws + 64880640);      // 62,914,560 B (NHWC128)
  _Float16* corrh = (_Float16*)(ws + 127795200);     // 31,457,280 B (NHWC64)
  _Float16* h1h   = (_Float16*)(ws + 159252480);     // 62,914,560 B (NHWC128)
  _Float16* w1p   = (_Float16*)(ws + 222167040);     //    147,456 B
  _Float16* w2p   = (_Float16*)(ws + 222314496);     //    147,456 B
  _Float16* w3p   = (_Float16*)(ws + 222461952);     //     36,864 B
  _Float16* h2h   = f1h;    // reuse: f1h dead after correlation  (NHWC64)
  _Float16* h3h   = f2wh;   // reuse: warped dead after correlation (NHWC32)

  const int pixTotal = Bn * Hn * Wn;                 // 245,760
  upflow_kernel<<<(Bn * 2 * Hn * Wn + 255) / 256, 256, 0, stream>>>(flow, up_w, flow_up);
  f1cvt_kernel<<<(pixTotal + 255) / 256, 256, 0, stream>>>(feats, f1h);
  warp_kernel<<<(pixTotal + 255) / 256, 256, 0, stream>>>(feats, flow_up, f2wh);
  pack_w_kernel<<<18, 256, 0, stream>>>(w1, w1p, 128, 49, 64);
  pack_w_kernel<<<18, 256, 0, stream>>>(w2, w2p, 64, 128, 128);
  pack_w_kernel<<<5, 256, 0, stream>>>(w3, w3p, 32, 64, 64);

  dim3 tiles(Wn / 16, Hn, Bn);
  corr_wmma<<<tiles, 64, 0, stream>>>(f1h, f2wh, corrh);
  conv3x3_wmma<64, 128><<<tiles, 256, 0, stream>>>(corrh, w1p, b1, h1h);
  conv3x3_wmma<128, 64><<<tiles, 128, 0, stream>>>(h1h, w2p, b2, h2h);
  conv3x3_wmma<64, 32><<<tiles, 64, 0, stream>>>(h2h, w3p, b3, h3h);
  conv4_flow<<<(pixTotal + 255) / 256, 256, 0, stream>>>(h3h, w4, b4, flow_up, out);
}
